// MultiGCNBlock_82858509074792
// MI455X (gfx1250) — compile-verified, matrix operand order checked by reading, not run on hardware
//
#include <hip/hip_runtime.h>
#include <hip/hip_bf16.h>
#include <stdint.h>

// ---------------------------------------------------------------------------
// MI455X (gfx1250) fused GNN block.
//   - bf16 WMMA (v_wmma_f32_16x16x32_bf16) for the big edge/message MLPs
//   - f32  WMMA (v_wmma_f32_16x16x4_f32)   for the small TAGConv GEMMs
// ---------------------------------------------------------------------------

typedef __attribute__((ext_vector_type(16))) __bf16 v16bf;
typedef __attribute__((ext_vector_type(8)))  float  v8f;
typedef __attribute__((ext_vector_type(2)))  float  v2f;

#define NN    100000
#define EE    300000
#define NFEATC 64
#define EFEATC 16
#define HID   128
#define OUTF  64

#define ROWS 128            // directed edges per workgroup tile
#define RS   264            // LDS row stride in bf16 (256 + 8 pad)

// fast SiLU: v * v_rcp_f32(1 + exp(-v))  (avoids precise-division expansion;
// TRANS ops co-execute with WMMA on CDNA5)
__device__ __forceinline__ float silu_f(float v) {
    return v * __builtin_amdgcn_rcpf(1.0f + __expf(-v));
}

// ---- A fragment (16x32 bf16) from LDS, ISA 16-bit A layout ----------------
// lane 0-15 : M=lane,    K = 0..7 (elems 0..7), 16..23 (elems 8..15)
// lane 16-31: M=lane-16, K = 8..15,            24..31
__device__ __forceinline__ v16bf lds_afrag(const __bf16* lds, int m0, int k0, int lane) {
    int m  = lane & 15;
    int kb = (lane & 16) ? 8 : 0;
    const __bf16* p = lds + (size_t)(m0 + m) * RS + k0 + kb;
    union { v16bf v; uint4 q[2]; } u;
    u.q[0] = *reinterpret_cast<const uint4*>(p);        // K = kb .. kb+7
    u.q[1] = *reinterpret_cast<const uint4*>(p + 16);   // K = kb+16 .. kb+23
    return u.v;
}

// ---- B fragment (32x16 bf16) from pre-packed global weights ---------------
// packed layout: [(kt*ntiles + nt)*32 + lane][16]  -> 32B contiguous per lane
__device__ __forceinline__ v16bf g_bfrag(const __bf16* __restrict__ wp, int ntiles,
                                         int kt, int nt, int lane) {
    const __bf16* p = wp + ((size_t)(kt * ntiles + nt) * 32 + lane) * 16;
    union { v16bf v; uint4 q[2]; } u;
    u.q[0] = *reinterpret_cast<const uint4*>(p);
    u.q[1] = *reinterpret_cast<const uint4*>(p + 8);
    return u.v;
}

// ---- GEMM stage: C(128 x ntB*16) = A(128 x KT*32) * B, epilogue -> LDS ----
__device__ __forceinline__
void gemm_stage(const __bf16* Alds, int KT,
                const __bf16* __restrict__ wp, int ntilesB, int CT,
                const float* __restrict__ bias, bool doSilu,
                __bf16* dstLds, int dstCol0,
                int w, int lane) {
    for (int ct = 0; ct < CT; ++ct) {
        int nt = w * CT + ct;
        v8f acc[8] = {};
        for (int kt = 0; kt < KT; ++kt) {
            // issue B (global) + all 8 A (LDS) loads first: independent regs,
            // one wait, then 8 back-to-back WMMAs on distinct accumulators.
            v16bf bf = g_bfrag(wp, ntilesB, kt, nt, lane);
            v16bf af[8];
#pragma unroll
            for (int rt = 0; rt < 8; ++rt) af[rt] = lds_afrag(Alds, rt * 16, kt * 32, lane);
#pragma unroll
            for (int rt = 0; rt < 8; ++rt) {
                acc[rt] = __builtin_amdgcn_wmma_f32_16x16x32_bf16(
                    false, af[rt], false, bf, (short)0, acc[rt], false, false);
            }
        }
        int n  = lane & 15;
        int mb = (lane & 16) ? 8 : 0;
        int col = nt * 16 + n;
        float bv = bias ? bias[col] : 0.0f;
#pragma unroll
        for (int rt = 0; rt < 8; ++rt) {
#pragma unroll
            for (int r = 0; r < 8; ++r) {
                int row = rt * 16 + r + mb;
                float v = acc[rt][r] + bv;
                if (doSilu) v = silu_f(v);
                dstLds[(size_t)row * RS + dstCol0 + col] = (__bf16)v;
            }
        }
    }
}

// ---- final GEMM stage with scatter-add epilogue ---------------------------
__device__ __forceinline__
void gemm_scatter(const __bf16* Alds, int KT,
                  const __bf16* __restrict__ wp, int ntilesB,
                  const float* __restrict__ bias,
                  float* __restrict__ agg, const int* tgt, int nd,
                  int w, int lane) {
    int nt = w;                        // output width 128 -> 1 coltile / wave
    v8f acc[8] = {};
    for (int kt = 0; kt < KT; ++kt) {
        v16bf bf = g_bfrag(wp, ntilesB, kt, nt, lane);
        v16bf af[8];
#pragma unroll
        for (int rt = 0; rt < 8; ++rt) af[rt] = lds_afrag(Alds, rt * 16, kt * 32, lane);
#pragma unroll
        for (int rt = 0; rt < 8; ++rt) {
            acc[rt] = __builtin_amdgcn_wmma_f32_16x16x32_bf16(
                false, af[rt], false, bf, (short)0, acc[rt], false, false);
        }
    }
    int n  = lane & 15;
    int mb = (lane & 16) ? 8 : 0;
    int col = nt * 16 + n;
    float bv = bias[col];
    for (int rt = 0; rt < 8; ++rt) {
        for (int r = 0; r < 8; ++r) {
            int m = rt * 16 + r + mb;
            if (m < nd) {
                atomicAdd(&agg[(size_t)tgt[m] * HID + col], acc[rt][r] + bv);
            }
        }
    }
}

// ---- fused edge-transform + message MLP + scatter -------------------------
__global__ __launch_bounds__(256)
void msg_kernel(const float* __restrict__ x, const int* __restrict__ ei,
                const float* __restrict__ eattr,
                const __bf16* __restrict__ etw1p, const __bf16* __restrict__ etw2p,
                const __bf16* __restrict__ eaw1p, const __bf16* __restrict__ eaw2p,
                const float* __restrict__ et_b1, const float* __restrict__ et_b2,
                const float* __restrict__ ea_b1, const float* __restrict__ ea_b2,
                float* __restrict__ agg) {
    extern __shared__ __bf16 smem[];
    __bf16* buf0 = smem;                    // 128 x 264 bf16
    __bf16* buf1 = smem + (size_t)ROWS * RS;
    int*    tgt  = (int*)(smem + (size_t)2 * ROWS * RS);

    const int tid  = threadIdx.x;
    const int w    = tid >> 5;
    const int lane = tid & 31;
    const long d0  = (long)blockIdx.x * ROWS;
    long rem = 2L * EE - d0;
    const int nd = (int)(rem < ROWS ? rem : ROWS);

    // ---- gather x rows + edge_attr into LDS (fp32 -> bf16) ----
    {
        int i = tid >> 1;           // row 0..127
        int half = tid & 1;
        long d = d0 + i;
        long dc = (d < 2L * EE) ? d : (2L * EE - 1);
        int e = (dc < EE) ? (int)dc : (int)(dc - EE);
        int s = ei[e], t = ei[EE + e];
        int r2 = (dc < EE) ? s : t;   // source node  (x_j)
        int c2 = (dc < EE) ? t : s;   // target node  (x_i)
        if (half == 0) tgt[i] = c2;

        const float* ep = eattr + (size_t)e * EFEATC + half * 8;
#pragma unroll
        for (int j = 0; j < 8; ++j) buf0[(size_t)i * RS + half * 8 + j] = (__bf16)ep[j];
#pragma unroll
        for (int j = 0; j < 8; ++j) buf0[(size_t)i * RS + 16 + half * 8 + j] = (__bf16)0.0f;

        const float* xc = x + (size_t)c2 * NFEATC + half * 32;
        const float* xr = x + (size_t)r2 * NFEATC + half * 32;
#pragma unroll
        for (int j = 0; j < 32; ++j) {
            buf1[(size_t)i * RS + half * 32 + j]      = (__bf16)xc[j];
            buf1[(size_t)i * RS + 64 + half * 32 + j] = (__bf16)xr[j];
        }
    }
    __syncthreads();
    // edge MLP 1: (128x32 zero-padded) * (32x128) + b1 -> SiLU -> buf0[:,128:256]
    gemm_stage(buf0, 1, etw1p, 8, 1, et_b1, true,  buf0, 128, w, lane);
    __syncthreads();
    // edge MLP 2: * (128x128) + b2 -> buf1[:,128:256]  (h_e slot of message)
    gemm_stage(buf0 + 128, 4, etw2p, 8, 1, et_b2, false, buf1, 128, w, lane);
    __syncthreads();
    // message MLP 1: (128x256) * (256x256) + b -> SiLU -> buf0 full
    gemm_stage(buf1, 8, eaw1p, 16, 2, ea_b1, true, buf0, 0, w, lane);
    __syncthreads();
    // message MLP 2: (128x256) * (256x128) + b -> atomic scatter-add to agg
    gemm_scatter(buf0, 8, eaw2p, 8, ea_b2, agg, tgt, nd, w, lane);
}

// ---- TAGConv GEMM: out(N x 64) (+)= h(N x 128) @ Wk, fp32 WMMA ------------
__global__ __launch_bounds__(256)
void tag_gemm_kernel(const float* __restrict__ h, const float* __restrict__ wp,
                     float* __restrict__ outacc, int accumulate) {
    int w = threadIdx.x >> 5, lane = threadIdx.x & 31;
    int rowbase = blockIdx.x * 128 + w * 16;
    int m  = lane & 15;
    int kb = (lane & 16) ? 2 : 0;
    int row  = rowbase + m;
    int rowc = row < NN ? row : NN - 1;
    const float* hr = h + (size_t)rowc * HID;
    v8f acc[4] = {};
    for (int kt = 0; kt < 32; ++kt) {
        const float* ap = hr + kt * 4 + kb;
        v2f a; a[0] = ap[0]; a[1] = ap[1];
        v2f b[4];
#pragma unroll
        for (int ct = 0; ct < 4; ++ct) {
            const float* bp = wp + ((size_t)(kt * 4 + ct) * 32 + lane) * 2;
            b[ct][0] = bp[0]; b[ct][1] = bp[1];
        }
#pragma unroll
        for (int ct = 0; ct < 4; ++ct) {
            acc[ct] = __builtin_amdgcn_wmma_f32_16x16x4_f32(
                false, a, false, b[ct], (short)0, acc[ct], false, false);
        }
    }
    int n  = lane & 15;
    int mb = (lane & 16) ? 8 : 0;
#pragma unroll
    for (int ct = 0; ct < 4; ++ct) {
        int col = ct * 16 + n;
#pragma unroll
        for (int r = 0; r < 8; ++r) {
            int rr = rowbase + r + mb;
            if (rr < NN) {
                size_t idx = (size_t)rr * OUTF + col;
                float v = acc[ct][r];
                outacc[idx] = accumulate ? (outacc[idx] + v) : v;
            }
        }
    }
}

// ---- sparse propagation: hdst[col] += norm[e] * hsrc[row] -----------------
__global__ void prop_kernel(const float* __restrict__ hsrc, float* __restrict__ hdst,
                            const int* __restrict__ ei, const float* __restrict__ nrm) {
    long tid = (long)blockIdx.x * blockDim.x + threadIdx.x;
    long e = tid >> 5;
    int  g = (int)(tid & 31);
    if (e >= EE) return;
    int r = ei[e], c = ei[EE + e];
    float nv = nrm[e];
    float4 v = *reinterpret_cast<const float4*>(hsrc + (size_t)r * HID + g * 4);
    float* dp = hdst + (size_t)c * HID + g * 4;
    atomicAdd(dp + 0, nv * v.x); atomicAdd(dp + 1, nv * v.y);
    atomicAdd(dp + 2, nv * v.z); atomicAdd(dp + 3, nv * v.w);
}

// ---- weight packing: fp32 (KxN) -> bf16 WMMA B-fragment layout ------------
__global__ void pack_bf16_kernel(const float* __restrict__ W, __bf16* __restrict__ out,
                                 int K, int N, int KT) {
    long idx = (long)blockIdx.x * blockDim.x + threadIdx.x;
    long total = (long)KT * (N / 16) * 512;
    if (idx >= total) return;
    int j    = (int)(idx & 15);
    int lane = (int)((idx >> 4) & 31);
    long rest = idx >> 9;
    int ntiles = N / 16;
    int nt = (int)(rest % ntiles);
    int kt = (int)(rest / ntiles);
    int kb = (lane & 16) ? 8 : 0;
    int Kk = kt * 32 + kb + ((j < 8) ? j : j + 8);
    int n  = nt * 16 + (lane & 15);
    float v = (Kk < K) ? W[(size_t)Kk * N + n] : 0.0f;
    out[idx] = (__bf16)v;
}

// ---- weight packing: fp32 (KxN) -> f32 16x16x4 B-fragment layout ----------
__global__ void pack_f32_kernel(const float* __restrict__ W, float* __restrict__ out,
                                int K, int N) {
    long idx = (long)blockIdx.x * blockDim.x + threadIdx.x;
    long total = (long)(K / 4) * (N / 16) * 64;
    if (idx >= total) return;
    int j    = (int)(idx & 1);
    int lane = (int)((idx >> 1) & 31);
    long rest = idx >> 6;
    int ntiles = N / 16;
    int nt = (int)(rest % ntiles);
    int kt = (int)(rest / ntiles);
    int kb = (lane & 16) ? 2 : 0;
    int Kk = kt * 4 + kb + j;
    int n  = nt * 16 + (lane & 15);
    out[idx] = W[(size_t)Kk * N + n];
}

// ---- small utility kernels ------------------------------------------------
__global__ void zero4_kernel(float4* p, long n4) {
    long i = (long)blockIdx.x * blockDim.x + threadIdx.x;
    if (i < n4) p[i] = make_float4(0.f, 0.f, 0.f, 0.f);
}
__global__ void zero_kernel(float* p, long n) {
    long i = (long)blockIdx.x * blockDim.x + threadIdx.x;
    if (i < n) p[i] = 0.0f;
}
__global__ void silu_kernel(float* p, long n) {
    long i = (long)blockIdx.x * blockDim.x + threadIdx.x;
    if (i < n) p[i] = silu_f(p[i]);
}
__global__ void deg_kernel(const int* __restrict__ ei, float* __restrict__ deg) {
    long i = (long)blockIdx.x * blockDim.x + threadIdx.x;
    if (i < EE) atomicAdd(&deg[ei[EE + i]], 1.0f);
}
__global__ void norm_kernel(const int* __restrict__ ei, const float* __restrict__ deg,
                            float* __restrict__ nrm) {
    long i = (long)blockIdx.x * blockDim.x + threadIdx.x;
    if (i >= EE) return;
    float dr = deg[ei[i]], dc = deg[ei[EE + i]];
    float a = dr > 0.0f ? __frsqrt_rn(dr) : 0.0f;
    float b = dc > 0.0f ? __frsqrt_rn(dc) : 0.0f;
    nrm[i] = a * b;
}
__global__ void select_kernel(const float* __restrict__ outacc, const float* __restrict__ tag_b,
                              const int* __restrict__ bt, float* __restrict__ outp, int b) {
    long i = (long)blockIdx.x * blockDim.x + threadIdx.x;
    if (i >= (long)NN * OUTF) return;
    int n = (int)(i / OUTF), o = (int)(i % OUTF);
    if (bt[n] == b) outp[i] = outacc[i] + tag_b[o];
}

// ---------------------------------------------------------------------------
extern "C" void kernel_launch(void* const* d_in, const int* in_sizes, int n_in,
                              void* d_out, int out_size, void* d_ws, size_t ws_size,
                              hipStream_t stream) {
    (void)in_sizes; (void)n_in; (void)out_size; (void)ws_size;
    const float* x     = (const float*)d_in[0];
    const int*   ei    = (const int*)d_in[1];
    const float* eattr = (const float*)d_in[2];
    const int*   bt    = (const int*)d_in[3];
    auto P = [&](int b, int i) -> const float* { return (const float*)d_in[4 + b * 10 + i]; };
    // param order per branch: et_w1, et_b1, et_w2, et_b2, ea_w1, ea_b1, ea_w2, ea_b2, tag_w, tag_b
    const int KH[3] = {3, 5, 7};   // K+1 hop matrices per branch

    char* ws = (char*)d_ws;
    size_t off = 0;
    auto carve = [&](size_t bytes) -> char* {
        char* p = ws + off; off = (off + bytes + 255) & ~(size_t)255; return p;
    };
    __bf16* etw1p[3]; __bf16* etw2p[3]; __bf16* eaw1p[3]; __bf16* eaw2p[3]; float* tagwp[3];
    for (int b = 0; b < 3; ++b) {
        etw1p[b] = (__bf16*)carve((size_t)1 * 8 * 512 * 2);
        etw2p[b] = (__bf16*)carve((size_t)4 * 8 * 512 * 2);
        eaw1p[b] = (__bf16*)carve((size_t)8 * 16 * 512 * 2);
        eaw2p[b] = (__bf16*)carve((size_t)8 * 8 * 512 * 2);
        tagwp[b] = (float*)carve((size_t)KH[b] * 32 * 4 * 64 * 4);
    }
    float* nrm    = (float*)carve((size_t)EE * 4);
    float* deg    = (float*)carve((size_t)NN * 4);
    float* agg    = (float*)carve((size_t)NN * HID * 4);
    float* tmp    = (float*)carve((size_t)NN * HID * 4);
    float* outacc = (float*)carve((size_t)NN * OUTF * 4);
    float* outp   = (float*)d_out;

    const int TB = 256;
    auto blocks = [&](long n) { return (int)((n + TB - 1) / TB); };

    zero_kernel<<<blocks(NN), TB, 0, stream>>>(deg, NN);
    zero4_kernel<<<blocks((long)NN * OUTF / 4), TB, 0, stream>>>((float4*)outp, (long)NN * OUTF / 4);
    deg_kernel<<<blocks(EE), TB, 0, stream>>>(ei, deg);
    norm_kernel<<<blocks(EE), TB, 0, stream>>>(ei, deg, nrm);

    for (int b = 0; b < 3; ++b) {
        pack_bf16_kernel<<<blocks(1L * 8 * 512), TB, 0, stream>>>(P(b, 0), etw1p[b], EFEATC, HID, 1);
        pack_bf16_kernel<<<blocks(4L * 8 * 512), TB, 0, stream>>>(P(b, 2), etw2p[b], HID, HID, 4);
        pack_bf16_kernel<<<blocks(8L * 16 * 512), TB, 0, stream>>>(P(b, 4), eaw1p[b], 2 * NFEATC + HID, 2 * HID, 8);
        pack_bf16_kernel<<<blocks(8L * 8 * 512), TB, 0, stream>>>(P(b, 6), eaw2p[b], 2 * HID, HID, 8);
        for (int k = 0; k < KH[b]; ++k)
            pack_f32_kernel<<<blocks(32L * 4 * 64), TB, 0, stream>>>(
                P(b, 8) + (size_t)k * HID * OUTF, tagwp[b] + (size_t)k * 32 * 4 * 64, HID, OUTF);
    }

    const size_t SMEM = (size_t)2 * ROWS * RS * 2 + ROWS * 4;  // two bf16 bufs + tgt[]
    const int mgrid = (int)((2L * EE + ROWS - 1) / ROWS);
    const int tgrid = (NN + 127) / 128;

    for (int b = 0; b < 3; ++b) {
        zero4_kernel<<<blocks((long)NN * HID / 4), TB, 0, stream>>>((float4*)agg, (long)NN * HID / 4);
        msg_kernel<<<mgrid, 256, SMEM, stream>>>(x, ei, eattr,
            etw1p[b], etw2p[b], eaw1p[b], eaw2p[b],
            P(b, 1), P(b, 3), P(b, 5), P(b, 7), agg);
        silu_kernel<<<blocks((long)NN * HID), TB, 0, stream>>>(agg, (long)NN * HID);
        tag_gemm_kernel<<<tgrid, 256, 0, stream>>>(agg, tagwp[b], outacc, 0);
        float* h0 = agg; float* h1 = tmp;
        for (int k = 1; k < KH[b]; ++k) {
            zero4_kernel<<<blocks((long)NN * HID / 4), TB, 0, stream>>>((float4*)h1, (long)NN * HID / 4);
            prop_kernel<<<blocks((long)EE * 32), TB, 0, stream>>>(h0, h1, ei, nrm);
            tag_gemm_kernel<<<tgrid, 256, 0, stream>>>(h1, tagwp[b] + (size_t)k * 32 * 4 * 64, outacc, 1);
            float* t = h0; h0 = h1; h1 = t;
        }
        select_kernel<<<blocks((long)NN * OUTF), TB, 0, stream>>>(outacc, P(b, 9), bt, outp, b);
    }
}
